// MMD_8100308320845
// MI455X (gfx1250) — matrix operand order repaired and checked by reading, not compiled, not run on hardware
//
#include <hip/hip_runtime.h>
#include <cstdint>
#include <cstddef>

// ---------------------------------------------------------------------------
// MMD loss, fused CDNA5 WMMA implementation (register-blocked).
//   B=8 batches, C=256 features, N=1024 samples/side, 2N=2048 rows per batch.
//   Gram = AT * AT^T per batch, AT = (2048 x 256) bf16 hi/lo split.
//   Each wave computes a 32x32 output region = 2x2 WMMA tiles:
//     per k-step: 16 x global_load_b128 feed 12 x v_wmma_f32_16x16x32_bf16.
// ---------------------------------------------------------------------------

typedef __attribute__((ext_vector_type(16))) __bf16 v16bf;
typedef __attribute__((ext_vector_type(8)))  float  v8f;

#define NB    8
#define NC    256
#define NHW   1024
#define NROW  2048   // 2N rows per batch

struct U32x8 { uint4 lo, hi; };

__device__ __forceinline__ uint4 ld128(const unsigned short* p) {
    return *reinterpret_cast<const uint4*>(p);
}
__device__ __forceinline__ v16bf ldfrag(const unsigned short* p0, const unsigned short* p1) {
    U32x8 q;
    q.lo = ld128(p0);
    q.hi = ld128(p1);
    return __builtin_bit_cast(v16bf, q);
}
__device__ __forceinline__ unsigned short bf16_rne(float x, float* hival) {
    uint32_t u = __builtin_bit_cast(uint32_t, x);
    uint32_t r = u + 0x7FFFu + ((u >> 16) & 1u);
    unsigned short h = (unsigned short)(r >> 16);
    uint32_t hb = ((uint32_t)h) << 16;
    *hival = __builtin_bit_cast(float, hb);
    return h;
}

// ---- scalars in ws: [1]=S2 (sum rowsum^2) [2]=bandwidth [3]=loss accum
__global__ void init_scalars_kernel(double* scal) {
    if (threadIdx.x < 8) scal[threadIdx.x] = 0.0;
}

// One block per (b, p): transpose column p of A into AT row p, split fp32 -> bf16 hi/lo,
// and compute exact fp32 row norm sq[b,p] via LDS reduction.
__global__ __launch_bounds__(256) void prep_kernel(
    const float* __restrict__ src, const float* __restrict__ tgt,
    unsigned short* __restrict__ ATh, unsigned short* __restrict__ ATl,
    float* __restrict__ sq)
{
    const int b = blockIdx.x >> 11;
    const int p = blockIdx.x & (NROW - 1);
    const int c = threadIdx.x;
    const float* base = (p < NHW) ? src : tgt;
    const int pl = p & (NHW - 1);
    float x = base[(((size_t)b * NC + c) << 10) + pl];

    float hf;
    unsigned short h = bf16_rne(x, &hf);
    float dummy;
    unsigned short l = bf16_rne(x - hf, &dummy);
    size_t idx = ((size_t)(b * NROW + p)) * NC + c;
    ATh[idx] = h;
    ATl[idx] = l;

    __shared__ float red[256];
    red[c] = x * x;
    __syncthreads();
    for (int off = 128; off > 0; off >>= 1) {
        if (c < off) red[c] += red[c + off];
        __syncthreads();
    }
    if (c == 0) sq[b * NROW + p] = red[0];
}

// One block per (b, c): rowsum over 2048 samples; accumulate rowsum^2 into scal[1].
__global__ __launch_bounds__(256) void rowsum_kernel(
    const float* __restrict__ src, const float* __restrict__ tgt, double* scal)
{
    const int b = blockIdx.x >> 8;
    const int c = blockIdx.x & (NC - 1);
    const float* s = src + (((size_t)b * NC + c) << 10);
    const float* t = tgt + (((size_t)b * NC + c) << 10);
    float acc = 0.f;
    #pragma unroll
    for (int j = 0; j < 4; ++j) {
        acc += s[threadIdx.x + 256 * j];
        acc += t[threadIdx.x + 256 * j];
    }
    __shared__ float red[256];
    red[threadIdx.x] = acc;
    __syncthreads();
    for (int off = 128; off > 0; off >>= 1) {
        if (threadIdx.x < off) red[threadIdx.x] += red[threadIdx.x + off];
        __syncthreads();
    }
    if (threadIdx.x == 0) {
        double r = (double)red[0];
        atomicAdd(&scal[1], r * r);
    }
}

// Single block: S1 = sum of all sq; bandwidth = (2n*S1 - 2*S2)/(n^2-n)/4
__global__ __launch_bounds__(256) void bandwidth_kernel(const float* __restrict__ sq, double* scal)
{
    double a = 0.0;
    for (int i = threadIdx.x; i < NB * NROW; i += 256) a += (double)sq[i];
    __shared__ double red[256];
    red[threadIdx.x] = a;
    __syncthreads();
    for (int off = 128; off > 0; off >>= 1) {
        if (threadIdx.x < off) red[threadIdx.x] += red[threadIdx.x + off];
        __syncthreads();
    }
    if (threadIdx.x == 0) {
        double S1 = red[0];
        double S2 = scal[1];
        double n  = (double)NROW;
        double sumL2 = 2.0 * n * S1 - 2.0 * S2;
        double bw = sumL2 / (n * n - n);
        bw *= 0.25;   // / KERNEL_MUL^(KERNEL_NUM//2) = / 2^2
        scal[2] = bw;
    }
}

__device__ __forceinline__ v8f wmma_bf16(v16bf a, v16bf b, v8f c) {
    return __builtin_amdgcn_wmma_f32_16x16x32_bf16(
        /*neg_a=*/false, a, /*neg_b=*/false, b,
        /*c_mod=*/(short)0, c, /*reuse_a=*/false, /*reuse_b=*/false);
}

// Each wave owns a 32x32 region (2x2 tiles of 16x16): K loop of 8 steps over C=256,
// 12 WMMAs / step (hi*hi + hi*lo + lo*hi for each tile).  Fused epilogue:
// L2 -> 5x v_exp_f32 -> signed wave/block reduction -> one f64 atomic per block.
__global__ __launch_bounds__(256) void mmd_main_kernel(
    const unsigned short* __restrict__ ATh, const unsigned short* __restrict__ ATl,
    const float* __restrict__ sq, const double* __restrict__ scal,
    double* __restrict__ lossAcc)
{
    const int wave = threadIdx.x >> 5;
    const int lane = threadIdx.x & 31;
    const int tile = blockIdx.x * 8 + wave;          // 32768 32x32 tiles total
    const int b    = tile >> 12;                     // 64*64 tiles per batch
    const int rem  = tile & 4095;
    const int p0   = (rem >> 6) << 5;
    const int q0   = (rem & 63) << 5;
    const int l15  = lane & 15;
    const int h16  = lane >> 4;

    const unsigned short* Ah = ATh + (size_t)b * NROW * NC;
    const unsigned short* Al = ATl + (size_t)b * NROW * NC;

    // ISA 16-bit A(16x32) layout: lane<16 -> K 0..7 & 16..23 ; lane>=16 -> K 8..15 & 24..31
    const size_t baseA0 = (size_t)(p0 + l15) * NC + (size_t)(h16 * 8);
    const size_t baseA1 = baseA0 + (size_t)16 * NC;
    // ISA 16-bit B(32x16) layout: lane<16 -> K 0..7 & 16..23 ; lane>=16 -> K 8..15 & 24..31
    const size_t baseB0 = (size_t)(q0 + l15) * NC + (size_t)(h16 * 8);
    const size_t baseB1 = baseB0 + (size_t)16 * NC;

    v8f acc[2][2] = {};
    #pragma unroll
    for (int kk = 0; kk < 8; ++kk) {
        const int c0 = kk * 32;
        v16bf a_hi[2], a_lo[2], b_hi[2], b_lo[2];
        a_hi[0] = ldfrag(Ah + baseA0 + c0, Ah + baseA0 + c0 + 16);
        a_lo[0] = ldfrag(Al + baseA0 + c0, Al + baseA0 + c0 + 16);
        a_hi[1] = ldfrag(Ah + baseA1 + c0, Ah + baseA1 + c0 + 16);
        a_lo[1] = ldfrag(Al + baseA1 + c0, Al + baseA1 + c0 + 16);
        b_hi[0] = ldfrag(Ah + baseB0 + c0, Ah + baseB0 + c0 + 16);
        b_lo[0] = ldfrag(Al + baseB0 + c0, Al + baseB0 + c0 + 16);
        b_hi[1] = ldfrag(Ah + baseB1 + c0, Ah + baseB1 + c0 + 16);
        b_lo[1] = ldfrag(Al + baseB1 + c0, Al + baseB1 + c0 + 16);
        #pragma unroll
        for (int i = 0; i < 2; ++i) {
            #pragma unroll
            for (int j = 0; j < 2; ++j) {
                acc[i][j] = wmma_bf16(a_hi[i], b_hi[j], acc[i][j]);
                acc[i][j] = wmma_bf16(a_hi[i], b_lo[j], acc[i][j]);
                acc[i][j] = wmma_bf16(a_lo[i], b_hi[j], acc[i][j]);
            }
        }
    }

    const float bw   = (float)scal[2];
    const float r0   = 1.0f / bw;
    const float* sqb = sq + b * NROW;
    const float L2E  = 1.44269504088896f;           // log2(e): v_exp_f32 is base-2

    float partial = 0.f;
    #pragma unroll
    for (int i = 0; i < 2; ++i) {
        float sqm[8];
        #pragma unroll
        for (int r = 0; r < 8; ++r) sqm[r] = sqb[p0 + 16 * i + r + h16 * 8];
        #pragma unroll
        for (int j = 0; j < 2; ++j) {
            const float sqn = sqb[q0 + 16 * j + l15];
            #pragma unroll
            for (int r = 0; r < 8; ++r) {
                const float g   = acc[i][j][r];
                const float L2v = sqm[r] + sqn - 2.0f * g;
                const float k   = -L2v * r0 * L2E;
                partial += __builtin_amdgcn_exp2f(k)
                         + __builtin_amdgcn_exp2f(k * 0.5f)
                         + __builtin_amdgcn_exp2f(k * 0.25f)
                         + __builtin_amdgcn_exp2f(k * 0.125f)
                         + __builtin_amdgcn_exp2f(k * 0.0625f);
            }
        }
    }

    // wave32 reduce
    for (int off = 16; off > 0; off >>= 1) partial += __shfl_xor(partial, off, 32);

    const float wsign = ((p0 < NHW) == (q0 < NHW)) ? 1.0f : -1.0f;  // XX,YY:+  XY,YX:-
    __shared__ float wp[8];
    if (lane == 0) wp[wave] = partial * wsign;
    __syncthreads();
    if (threadIdx.x == 0) {
        float s = 0.f;
        #pragma unroll
        for (int i = 0; i < 8; ++i) s += wp[i];
        atomicAdd(lossAcc, (double)s);
    }
}

__global__ void finalize_kernel(const double* scal, float* out) {
    if (threadIdx.x == 0)
        out[0] = (float)(scal[3] / ((double)NB * (double)NHW * (double)NHW));
}

// ---------------------------------------------------------------------------
extern "C" void kernel_launch(void* const* d_in, const int* in_sizes, int n_in,
                              void* d_out, int out_size, void* d_ws, size_t ws_size,
                              hipStream_t stream) {
    const float* src = (const float*)d_in[0];
    const float* tgt = (const float*)d_in[1];
    float* out = (float*)d_out;

    // workspace layout
    char* base = (char*)d_ws;
    double* scal = (double*)base;                                  // 8 doubles
    float*  sq   = (float*)(base + 256);                           // 8*2048 floats = 64 KB
    unsigned short* ATh = (unsigned short*)(base + 256 + 65536);   // 8 MB
    unsigned short* ATl = ATh + (size_t)NB * NROW * NC;            // 8 MB

    init_scalars_kernel<<<1, 64, 0, stream>>>(scal);
    prep_kernel<<<NB * NROW, 256, 0, stream>>>(src, tgt, ATh, ATl, sq);
    rowsum_kernel<<<NB * NC, 256, 0, stream>>>(src, tgt, scal);
    bandwidth_kernel<<<1, 256, 0, stream>>>(sq, scal);
    // 8 * 64 * 64 = 32768 wave-tiles of 32x32, 8 waves per block
    mmd_main_kernel<<<(NB * 64 * 64) / 8, 256, 0, stream>>>(ATh, ATl, sq, scal, &scal[3]);
    finalize_kernel<<<1, 32, 0, stream>>>(scal, out);
}
